// GCU_33165737459948
// MI455X (gfx1250) — matrix operand order repaired and checked by reading, not compiled, not run on hardware
//
#include <hip/hip_runtime.h>

typedef __attribute__((ext_vector_type(16))) __bf16 v16bf;
typedef __attribute__((ext_vector_type(8)))  __bf16 v8bf;
typedef __attribute__((ext_vector_type(8)))  float  v8f;

#define DEV __device__ __forceinline__

DEV unsigned short f2bf(float f) {
  unsigned u = __float_as_uint(f);
  unsigned r = u + 0x7FFFu + ((u >> 16) & 1u);   // round-to-nearest-even
  return (unsigned short)(r >> 16);
}
DEV unsigned pack2(float a, float b) {
  return (unsigned)f2bf(a) | ((unsigned)f2bf(b) << 16);
}
DEV v8f wmma_bf16(v16bf a, v16bf b, v8f c) {
  return __builtin_amdgcn_wmma_f32_16x16x32_bf16(false, a, false, b, (short)0, c, false, false);
}
// Build a 16x32 bf16 A-fragment from a row-major bf16 tile in LDS.
// Lane l: row = l&15, half = l>>4; k = 16*half + s, one contiguous 32B run per chunk.
DEV v16bf lds_fragA(const unsigned short* tile, int lane, int chunk, int strideB) {
  const char* p = (const char*)tile + (lane & 15) * strideB + chunk * 64 + (lane >> 4) * 32;
  v8bf lo = *(const v8bf*)p;
  v8bf hi = *(const v8bf*)(p + 16);
  return __builtin_shufflevector(lo, hi, 0,1,2,3,4,5,6,7,8,9,10,11,12,13,14,15);
}
DEV void lds_fence() { asm volatile("s_wait_dscnt 0" ::: "memory"); }

// ---------------------------------------------------------------- utility
__global__ void zero_kernel(float* p, size_t n) {
  size_t i = (size_t)blockIdx.x * blockDim.x + threadIdx.x;
  size_t s = (size_t)gridDim.x * blockDim.x;
  for (; i < n; i += s) p[i] = 0.f;
}
__global__ void cvt_kernel(const float* __restrict__ src, unsigned short* __restrict__ dst, int n) {
  int i = blockIdx.x * blockDim.x + threadIdx.x;
  int s = gridDim.x * blockDim.x;
  for (; i < n; i += s) dst[i] = f2bf(src[i]);
}
__global__ void finalize_bn(const float* __restrict__ sum, const float* __restrict__ sq,
                            const float* __restrict__ g, const float* __restrict__ be,
                            float* __restrict__ sOut, float* __restrict__ tOut,
                            float cnt, int H) {
  int c = blockIdx.x * blockDim.x + threadIdx.x;
  if (c < H) {
    float mu  = sum[c] / cnt;
    float var = sq[c] / cnt - mu * mu;
    float s   = g[c] * rsqrtf(var + 1e-5f);
    sOut[c] = s;
    tOut[c] = be[c] - mu * s;
  }
}
// Fold BN2 affines (per concat channel) into the final 128x128 weights/bias.
__global__ void fold_final(const float* __restrict__ Wf, const float* __restrict__ bfin,
                           const float* __restrict__ s2t, const float* __restrict__ t2t,
                           const float* __restrict__ s2g, const float* __restrict__ t2g,
                           unsigned short* __restrict__ Wout, float* __restrict__ bout) {
  __shared__ float red[128];
  int n = blockIdx.x, k = threadIdx.x;
  float s = (k < 64) ? s2t[k] : s2g[k - 64];
  float t = (k < 64) ? t2t[k] : t2g[k - 64];
  float w = Wf[n * 128 + k];
  Wout[n * 128 + k] = f2bf(w * s);
  red[k] = w * t;
  __syncthreads();
  for (int o = 64; o > 0; o >>= 1) { if (k < o) red[k] += red[k + o]; __syncthreads(); }
  if (k == 0) bout[n] = bfin[n] + red[0];
}

// ---------------------------------------------------------------- edge pass
// PASSB=0: gather + GEMM1 + ReLU, accumulate BN1 stats.
// PASSB=1: gather + GEMM1 + ReLU + BN1-affine + GEMM2 + ReLU,
//          accumulate BN2 stats, atomic-umax scatter of h2 (>=0) to agg.
template<int PASSB>
__global__ __launch_bounds__(256) void edge_pass(
    const float* __restrict__ x,
    const int* __restrict__ srcI, const int* __restrict__ dstI,
    int E, int rows,
    const unsigned short* __restrict__ W1, const float* __restrict__ b1,
    const float* __restrict__ s1, const float* __restrict__ t1,
    const unsigned short* __restrict__ W2, const float* __restrict__ b2,
    float* __restrict__ sumO, float* __restrict__ sqO,
    unsigned* __restrict__ agg)
{
  __shared__ __align__(16) unsigned short smem[8 * 2080];   // per-wave: msg(2KB)+act(2KB)+dst(64B)
  const int lane = threadIdx.x & 31;
  // wave index is uniform within a wave: pin it to an SGPR so tile loop,
  // full-tile test and LDS bases stay scalar.
  const int wid  = __builtin_amdgcn_readfirstlane(threadIdx.x >> 5);
  const int gw   = blockIdx.x * 8 + wid;
  const int nw   = gridDim.x * 8;
  unsigned short* msg = smem + wid * 2080;
  unsigned short* act = msg + 1024;
  int* dlds = (int*)(msg + 2048);

  const int row = lane & 15;
  const int h   = lane >> 4;
  const int col = lane & 15;
  const int m0  = h * 8;

  v16bf B1[4][2], B2[4][2];
  float b1c[4], b2c[4], s1c[4], t1c[4];
#pragma unroll
  for (int t = 0; t < 4; ++t) {
    int n = 16 * t + col;
#pragma unroll
    for (int c = 0; c < 2; ++c) {
      B1[t][c] = *(const v16bf*)(W1 + n * 64 + 32 * c + 16 * h);
      if (PASSB) B2[t][c] = *(const v16bf*)(W2 + n * 64 + 32 * c + 16 * h);
    }
    b1c[t] = b1[n];
    if (PASSB) { b2c[t] = b2[n]; s1c[t] = s1[n]; t1c[t] = t1[n]; }
  }

  float sum[4] = {0,0,0,0}, sq[4] = {0,0,0,0};
  const int T = (rows + 15) >> 4;

  for (int tile = gw; tile < T; tile += nw) {
    const bool full = (tile * 16 + 16) <= rows;   // scalar (tile is SGPR-derived)
    { // gather 16 edges -> bf16 message tile [16 rows x 64 ch], row-major
      int r = tile * 16 + row;
      if (r >= rows) r = rows - 1;
      int si, di;
      if (r < E) { si = srcI[r]; di = dstI[r]; }
      else       { si = di = r - E; }              // self loops
      const float4* xi4 = (const float4*)(x + (size_t)di * 32) + h * 4;
      const float4* xj4 = (const float4*)(x + (size_t)si * 32) + h * 4;
      unsigned o0[8], o1[8];
#pragma unroll
      for (int j = 0; j < 4; ++j) {
        float4 a  = xi4[j];
        float4 bb = xj4[j];
        o0[2*j]   = pack2(a.x, a.y);
        o0[2*j+1] = pack2(a.z, a.w);
        o1[2*j]   = pack2(bb.x - a.x, bb.y - a.y);
        o1[2*j+1] = pack2(bb.z - a.z, bb.w - a.w);
      }
      uint4* p0 = (uint4*)((char*)msg + row * 128 + 32 * h);
      p0[0] = make_uint4(o0[0], o0[1], o0[2], o0[3]);
      p0[1] = make_uint4(o0[4], o0[5], o0[6], o0[7]);
      uint4* p1 = (uint4*)((char*)msg + row * 128 + 64 + 32 * h);
      p1[0] = make_uint4(o1[0], o1[1], o1[2], o1[3]);
      p1[1] = make_uint4(o1[4], o1[5], o1[6], o1[7]);
      if (h == 0) dlds[row] = di;
    }
    lds_fence();

    v16bf A0 = lds_fragA(msg, lane, 0, 128);
    v16bf A1 = lds_fragA(msg, lane, 1, 128);

    v8f acc[4];
#pragma unroll
    for (int t = 0; t < 4; ++t) {
      v8f z = {0,0,0,0,0,0,0,0};
      z = wmma_bf16(A0, B1[t][0], z);
      z = wmma_bf16(A1, B1[t][1], z);
      acc[t] = z;
    }

    if (!PASSB) {
      if (full) {
#pragma unroll
        for (int t = 0; t < 4; ++t)
#pragma unroll
          for (int r = 0; r < 8; ++r) {
            float v = acc[t][r] + b1c[t];
            v = v > 0.f ? v : 0.f;
            sum[t] += v; sq[t] += v * v;
          }
      } else {
#pragma unroll
        for (int t = 0; t < 4; ++t)
#pragma unroll
          for (int r = 0; r < 8; ++r) {
            float v = acc[t][r] + b1c[t];
            v = v > 0.f ? v : 0.f;
            if (tile * 16 + m0 + r < rows) { sum[t] += v; sq[t] += v * v; }
          }
      }
    } else {
#pragma unroll
      for (int t = 0; t < 4; ++t)
#pragma unroll
        for (int r = 0; r < 8; ++r) {
          float v = acc[t][r] + b1c[t];
          v = v > 0.f ? v : 0.f;
          v = v * s1c[t] + t1c[t];                  // BN1 affine
          act[(m0 + r) * 64 + 16 * t + col] = f2bf(v);
        }
      lds_fence();
      v16bf C0 = lds_fragA(act, lane, 0, 128);
      v16bf C1 = lds_fragA(act, lane, 1, 128);
      int dbase[8];
#pragma unroll
      for (int r = 0; r < 8; ++r) dbase[r] = dlds[m0 + r];  // hoisted, reused across t
      v8f acc2[4];
#pragma unroll
      for (int t = 0; t < 4; ++t) {
        v8f z = {0,0,0,0,0,0,0,0};
        z = wmma_bf16(C0, B2[t][0], z);
        z = wmma_bf16(C1, B2[t][1], z);
        acc2[t] = z;
      }
      if (full) {
#pragma unroll
        for (int t = 0; t < 4; ++t)
#pragma unroll
          for (int r = 0; r < 8; ++r) {
            float v = acc2[t][r] + b2c[t];
            v = v > 0.f ? v : 0.f;                  // h2 >= 0
            sum[t] += v; sq[t] += v * v;
            atomicMax(&agg[(size_t)dbase[r] * 64 + 16 * t + col], __float_as_uint(v));
          }
      } else {
#pragma unroll
        for (int t = 0; t < 4; ++t)
#pragma unroll
          for (int r = 0; r < 8; ++r) {
            float v = acc2[t][r] + b2c[t];
            v = v > 0.f ? v : 0.f;
            if (tile * 16 + m0 + r < rows) {
              sum[t] += v; sq[t] += v * v;
              atomicMax(&agg[(size_t)dbase[r] * 64 + 16 * t + col], __float_as_uint(v));
            }
          }
      }
    }
  }
#pragma unroll
  for (int t = 0; t < 4; ++t) {
    atomicAdd(&sumO[16 * t + col], sum[t]);
    atomicAdd(&sqO[16 * t + col], sq[t]);
  }
}

// ---------------------------------------------------------------- node pass
// FINAL=0: GEMM(agg_concat -> 128) + ReLU, accumulate final BN stats.
// FINAL=1: same GEMM, apply final BN affine, write d_out.
template<int FINAL>
__global__ __launch_bounds__(256) void node_pass(
    const unsigned* __restrict__ aggT, const unsigned* __restrict__ aggG,
    const unsigned short* __restrict__ Wf, const float* __restrict__ bfb,
    float* __restrict__ sumO, float* __restrict__ sqO,
    const float* __restrict__ sF, const float* __restrict__ tF,
    float* __restrict__ out, int N)
{
  __shared__ __align__(16) unsigned short smem[8 * 2048];   // per-wave 16x128 bf16
  const int lane = threadIdx.x & 31;
  const int wid  = __builtin_amdgcn_readfirstlane(threadIdx.x >> 5);
  const int gw = blockIdx.x * 8 + wid, nw = gridDim.x * 8;
  unsigned short* yt = smem + wid * 2048;
  const int row = lane & 15, h = lane >> 4, col = lane & 15, m0 = 8 * h;

  v16bf B[8][4];
  float bc[8], sc[8], tc[8];
#pragma unroll
  for (int t = 0; t < 8; ++t) {
    int n = 16 * t + col;
#pragma unroll
    for (int c = 0; c < 4; ++c)
      B[t][c] = *(const v16bf*)(Wf + n * 128 + 32 * c + 16 * h);
    bc[t] = bfb[n];
    if (FINAL) { sc[t] = sF[n]; tc[t] = tF[n]; }
  }
  float sum[8] = {0,0,0,0,0,0,0,0}, sq[8] = {0,0,0,0,0,0,0,0};
  const unsigned* aggp = h ? aggG : aggT;    // half 0: tpl channels 0-63, half 1: geo 64-127
  const int T = (N + 15) >> 4;

  for (int tile = gw; tile < T; tile += nw) {
    const bool full = (tile * 16 + 16) <= N;  // scalar
    int node = tile * 16 + row;
    int nc = node < N ? node : N - 1;
    const uint4* ar = (const uint4*)(aggp + (size_t)nc * 64);
    unsigned pk[32];
#pragma unroll
    for (int q = 0; q < 16; ++q) {
      uint4 u = ar[q];                         // umax bit patterns of floats >= 0
      pk[2*q]   = pack2(__uint_as_float(u.x), __uint_as_float(u.y));
      pk[2*q+1] = pack2(__uint_as_float(u.z), __uint_as_float(u.w));
    }
    uint4* dp = (uint4*)((char*)yt + row * 256 + 128 * h);
#pragma unroll
    for (int q = 0; q < 8; ++q)
      dp[q] = make_uint4(pk[4*q], pk[4*q+1], pk[4*q+2], pk[4*q+3]);
    lds_fence();
    v16bf A[4];
#pragma unroll
    for (int c = 0; c < 4; ++c) A[c] = lds_fragA(yt, lane, c, 256);
#pragma unroll
    for (int t = 0; t < 8; ++t) {
      v8f z = {0,0,0,0,0,0,0,0};
#pragma unroll
      for (int c = 0; c < 4; ++c) z = wmma_bf16(A[c], B[t][c], z);
      if (full) {
#pragma unroll
        for (int r = 0; r < 8; ++r) {
          float v = z[r] + bc[t];
          v = v > 0.f ? v : 0.f;
          int nr = tile * 16 + m0 + r;
          if (FINAL) out[(size_t)nr * 128 + 16 * t + col] = v * sc[t] + tc[t];
          else { sum[t] += v; sq[t] += v * v; }
        }
      } else {
#pragma unroll
        for (int r = 0; r < 8; ++r) {
          float v = z[r] + bc[t];
          v = v > 0.f ? v : 0.f;
          int nr = tile * 16 + m0 + r;
          if (nr < N) {
            if (FINAL) out[(size_t)nr * 128 + 16 * t + col] = v * sc[t] + tc[t];
            else { sum[t] += v; sq[t] += v * v; }
          }
        }
      }
    }
  }
  if (!FINAL) {
#pragma unroll
    for (int t = 0; t < 8; ++t) {
      atomicAdd(&sumO[16 * t + col], sum[t]);
      atomicAdd(&sqO[16 * t + col], sq[t]);
    }
  }
}

// ---------------------------------------------------------------- launch
extern "C" void kernel_launch(void* const* d_in, const int* in_sizes, int n_in,
                              void* d_out, int out_size, void* d_ws, size_t ws_size,
                              hipStream_t stream) {
  const float* x = (const float*)d_in[0];
  const int N = in_sizes[0] / 32;
  const int E = in_sizes[1] / 2;
  const int rows = E + N;
  char* ws = (char*)d_ws;
  // ws layout: [0..64KB) bf16 weights (tplW1,tplW2,geoW1,geoW2 @0/8K/16K/24K, foldedWf @32K)
  //            [64KB..72KB) float stats/affines, [72KB..) agg buffers (2 * N * 64 f32)
  float* S = (float*)(ws + 65536);
  unsigned* agg0 = (unsigned*)(ws + 73728);
  unsigned* agg1 = agg0 + (size_t)N * 64;

  size_t zn = 2048 + (size_t)2 * N * 64;       // stats region + both agg buffers (contiguous)
  zero_kernel<<<2048, 256, 0, stream>>>(S, zn);

  cvt_kernel<<<16, 256, 0, stream>>>((const float*)d_in[3],  (unsigned short*)(ws + 0),     4096);
  cvt_kernel<<<16, 256, 0, stream>>>((const float*)d_in[7],  (unsigned short*)(ws + 8192),  4096);
  cvt_kernel<<<16, 256, 0, stream>>>((const float*)d_in[11], (unsigned short*)(ws + 16384), 4096);
  cvt_kernel<<<16, 256, 0, stream>>>((const float*)d_in[15], (unsigned short*)(ws + 24576), 4096);

  for (int g = 0; g < 2; ++g) {
    const int* ei        = (const int*)d_in[1 + g];
    const float* b1      = (const float*)d_in[4 + 8*g];
    const float* g1      = (const float*)d_in[5 + 8*g];
    const float* be1     = (const float*)d_in[6 + 8*g];
    const float* b2      = (const float*)d_in[8 + 8*g];
    const float* g2      = (const float*)d_in[9 + 8*g];
    const float* be2     = (const float*)d_in[10 + 8*g];
    const unsigned short* w1 = (const unsigned short*)(ws + g * 16384);
    const unsigned short* w2 = (const unsigned short*)(ws + g * 16384 + 8192);
    float* Sg = S + g * 512;   // sum1,sq1,sum2,sq2,s1,t1,s2,t2 (64 floats each)
    unsigned* agg = g ? agg1 : agg0;

    edge_pass<0><<<2048, 256, 0, stream>>>(x, ei, ei + E, E, rows, w1, b1,
        nullptr, nullptr, nullptr, nullptr, Sg + 0, Sg + 64, nullptr);
    finalize_bn<<<1, 64, 0, stream>>>(Sg + 0, Sg + 64, g1, be1, Sg + 256, Sg + 320,
        (float)rows, 64);
    edge_pass<1><<<2048, 256, 0, stream>>>(x, ei, ei + E, E, rows, w1, b1,
        Sg + 256, Sg + 320, w2, b2, Sg + 128, Sg + 192, agg);
    finalize_bn<<<1, 64, 0, stream>>>(Sg + 128, Sg + 192, g2, be2, Sg + 384, Sg + 448,
        (float)rows, 64);
  }

  // fold BN2 affines into the final MLP weights/bias (BN2 commutes with segment_max)
  fold_final<<<128, 128, 0, stream>>>((const float*)d_in[19], (const float*)d_in[20],
      S + 384, S + 448, S + 896, S + 960,
      (unsigned short*)(ws + 32768), S + 1536);

  node_pass<0><<<512, 256, 0, stream>>>(agg0, agg1, (const unsigned short*)(ws + 32768), S + 1536,
      S + 1024, S + 1152, nullptr, nullptr, nullptr, N);
  finalize_bn<<<1, 128, 0, stream>>>(S + 1024, S + 1152,
      (const float*)d_in[21], (const float*)d_in[22], S + 1280, S + 1408, (float)N, 128);
  node_pass<1><<<512, 256, 0, stream>>>(agg0, agg1, (const unsigned short*)(ws + 32768), S + 1536,
      nullptr, nullptr, S + 1280, S + 1408, (float*)d_out, N);

  (void)n_in; (void)out_size; (void)ws_size;
}
